// GPSGraph_7567732375849
// MI455X (gfx1250) — matrix operand order repaired and checked by reading, not compiled, not run on hardware
//
#include <hip/hip_runtime.h>
#include <hip/hip_bf16.h>

// ---------------------------------------------------------------------------
// GPSConv graph network for MI455X (gfx1250, wave32, WMMA).
// All dense matmuls + attention use v_wmma_f32_16x16x32_f16 (f16 in, f32 acc).
// ---------------------------------------------------------------------------

typedef __attribute__((ext_vector_type(16))) _Float16 v16h;
typedef __attribute__((ext_vector_type(8)))  float    v8f;

#define GN    4096      // nodes
#define GC    96        // channels
#define GE    131072    // edges
#define GHEADS 4
#define GDH   24        // head dim (padded to 32 for WMMA)

static __device__ __forceinline__ v8f wmma_f16(v16h a, v16h b, v8f c) {
  return __builtin_amdgcn_wmma_f32_16x16x32_f16(
      /*neg_a=*/false, a, /*neg_b=*/false, b,
      /*c_mod=*/(short)0, c, /*reuse_a=*/false, /*reuse_b=*/false);
}

static __device__ __forceinline__ float4 ld4(const float* p) {
  return *reinterpret_cast<const float4*>(p);
}

static __device__ __forceinline__ void pack8(v16h& v, int base, float4 x, float4 y) {
  v[base + 0] = (_Float16)x.x; v[base + 1] = (_Float16)x.y;
  v[base + 2] = (_Float16)x.z; v[base + 3] = (_Float16)x.w;
  v[base + 4] = (_Float16)y.x; v[base + 5] = (_Float16)y.y;
  v[base + 6] = (_Float16)y.z; v[base + 7] = (_Float16)y.w;
}

static __device__ __forceinline__ void pack8s(v16h& v, int base, float4 x, float4 y, float s) {
  v[base + 0] = (_Float16)(x.x * s); v[base + 1] = (_Float16)(x.y * s);
  v[base + 2] = (_Float16)(x.z * s); v[base + 3] = (_Float16)(x.w * s);
  v[base + 4] = (_Float16)(y.x * s); v[base + 5] = (_Float16)(y.y * s);
  v[base + 6] = (_Float16)(y.z * s); v[base + 7] = (_Float16)(y.w * s);
}

// A-matrix 16x32 f16 operand, contiguous k order used here: lane l holds row (l&15);
// element e -> k = ((l>=16)?16:0) + e  == one contiguous run.
static __device__ __forceinline__ v16h load_a16x32(const float* __restrict__ A,
                                                   int lda, int row0, int k0) {
  int lane = threadIdx.x & 31;
  int half = lane >> 4;
  const float* rowp = A + (size_t)(row0 + (lane & 15)) * lda + k0 + (half << 4);
  v16h a;
  pack8(a, 0, ld4(rowp), ld4(rowp + 4));
  pack8(a, 8, ld4(rowp + 8), ld4(rowp + 12));
  return a;
}

// B-matrix 32x16 f16 layout: lane l holds column (l&15);
// element e -> k = e + ((l>=16)?16:0)
static __device__ __forceinline__ v16h load_b32x16(const float* __restrict__ B,
                                                   int ldb, int k0, int col0) {
  int lane = threadIdx.x & 31;
  const float* colp = B + (size_t)(k0 + ((lane >> 4) << 4)) * ldb + col0 + (lane & 15);
  v16h b;
#pragma unroll
  for (int e = 0; e < 16; ++e) b[e] = (_Float16)colp[(size_t)e * ldb];
  return b;
}

// ---------------------------------------------------------------------------
// Generic WMMA GEMM: C[M,N] = act(A[M,K] @ B[K,N] + bias) (+ residual)
// 128 threads = 4 waves; wave w owns rows [blk.x*64 + w*16, +16), cols blk.y*16.
// M mult of 64, N mult of 16, K mult of 32.
// ---------------------------------------------------------------------------
template <bool RELU, bool HASRES>
__global__ void gemm_wmma_kernel(const float* __restrict__ A,
                                 const float* __restrict__ B,
                                 const float* __restrict__ bias,
                                 const float* __restrict__ residual,
                                 float* __restrict__ C,
                                 int M, int K, int N) {
  const int wid = threadIdx.x >> 5;
  const int tm = blockIdx.x * 64 + wid * 16;
  const int tn = blockIdx.y * 16;
  v8f acc = {};
  for (int k0 = 0; k0 < K; k0 += 32) {
    __builtin_prefetch(A + (size_t)(tm + (threadIdx.x & 15)) * K + k0 + 32, 0, 0);
    v16h a = load_a16x32(A, K, tm, k0);
    v16h b = load_b32x16(B, N, k0, tn);
    acc = wmma_f16(a, b, acc);
  }
  const int lane = threadIdx.x & 31;
  const int col  = tn + (lane & 15);
  const int rbase = tm + ((lane >> 4) << 3);
  const float bv = bias[col];
#pragma unroll
  for (int e = 0; e < 8; ++e) {
    int row = rbase + e;
    float v = acc[e] + bv;
    if (RELU) v = fmaxf(v, 0.f);
    if (HASRES) v += residual[(size_t)row * N + col];
    C[(size_t)row * N + col] = v;
  }
}

// ---------------------------------------------------------------------------
// Flash-style attention over all N nodes. One wave per (16-query tile, head).
// Q pre-scaled by 1/sqrt(dh); KV tiles of 32; online softmax; acc in f32.
// qkv: [N, 3C] with q|k|v each [N, HEADS, DH] packed in C columns.
// Zero-padding lives on the Q side (k=24..31), so K-tile loads are branchless;
// the second P.V accumulator's columns for d>=24 are computed-but-discarded,
// so V-tile loads are branchless too (any in-bounds garbage is never stored).
// ---------------------------------------------------------------------------
__global__ void attention_kernel(const float* __restrict__ qkv,
                                 float* __restrict__ out) {
  const int lane = threadIdx.x & 31;
  const int half = lane >> 4;
  const int hh   = blockIdx.y;
  const int q0   = blockIdx.x * 16;
  const int LDQ  = 3 * GC;
  const float scale = 0.2041241452319315f;  // 1/sqrt(24)
  const float4 f4z = {0.f, 0.f, 0.f, 0.f};

  __shared__ _Float16 pLDS[16][32];

  // Q tile as A-matrix (16 x 32, dh 24 padded with zeros), scale folded in.
  v16h qa;
  {
    const float* qrow = qkv + (size_t)(q0 + (lane & 15)) * LDQ + hh * GDH;
    float4 a0 = ld4(qrow + (half << 4));
    float4 a1 = ld4(qrow + (half << 4) + 4);
    float4 b0 = f4z, b1 = f4z;
    if (half == 0) { b0 = ld4(qrow + 8); b1 = ld4(qrow + 12); }
    pack8s(qa, 0, a0, a1, scale);
    pack8s(qa, 8, b0, b1, scale);
  }

  float mrow[8], lrow[8];
#pragma unroll
  for (int e = 0; e < 8; ++e) { mrow[e] = -1e30f; lrow[e] = 0.f; }
  v8f acc0 = {}, acc1 = {};

  for (int kv0 = 0; kv0 < GN; kv0 += 32) {
    // K tiles as B-matrices: B[d, key]; kb0 keys kv0..+15, kb1 +16..31.
    // Branchless: lane half h loads 16 contiguous floats at d = h*16; the
    // d=24..31 positions multiply against Q's zero padding.
    v16h kb0, kb1;
    {
      const float* kA = qkv + (size_t)(kv0 + (lane & 15)) * LDQ + GC + hh * GDH + (half << 4);
      const float* kB = kA + (size_t)16 * LDQ;
      pack8(kb0, 0, ld4(kA), ld4(kA + 4));
      pack8(kb0, 8, ld4(kA + 8), ld4(kA + 12));
      pack8(kb1, 0, ld4(kB), ld4(kB + 4));
      pack8(kb1, 8, ld4(kB + 8), ld4(kB + 12));
    }
    v8f s0 = {}, s1 = {};
    s0 = wmma_f16(qa, kb0, s0);
    s1 = wmma_f16(qa, kb1, s1);

    // Online softmax per query row (row = e + 8*half, key col = lane&15 / +16).
#pragma unroll
    for (int e = 0; e < 8; ++e) {
      float tmax = fmaxf(s0[e], s1[e]);
#pragma unroll
      for (int m = 1; m <= 8; m <<= 1) tmax = fmaxf(tmax, __shfl_xor(tmax, m, 32));
      float mnew = fmaxf(mrow[e], tmax);
      float corr = __expf(mrow[e] - mnew);
      float p0 = __expf(s0[e] - mnew);
      float p1 = __expf(s1[e] - mnew);
      float rs = p0 + p1;
#pragma unroll
      for (int m = 1; m <= 8; m <<= 1) rs += __shfl_xor(rs, m, 32);
      lrow[e] = lrow[e] * corr + rs;
      mrow[e] = mnew;
      acc0[e] *= corr;
      acc1[e] *= corr;
      int rw = e + (half << 3);
      pLDS[rw][lane & 15]        = (_Float16)p0;
      pLDS[rw][(lane & 15) + 16] = (_Float16)p1;
    }
    __syncthreads();

    // P (16x32) as A-matrix from LDS.
    v16h pa;
    {
      int r = lane & 15;
#pragma unroll
      for (int e = 0; e < 16; ++e) {
        int k = (half << 4) + e;
        pa[e] = pLDS[r][k];
      }
    }
    __syncthreads();

    // V tiles as B-matrices: B[key, d]; vb0 d 0..15, vb1 d 16..31.
    // Branchless: vb1 columns with d>=24 produce discarded outputs.
    v16h vb0, vb1;
    {
      const float* vbase = qkv + (size_t)(kv0 + (half << 4)) * LDQ + 2 * GC + hh * GDH;
      int d0 = lane & 15;
      int d1 = 16 + (lane & 15);
#pragma unroll
      for (int e = 0; e < 16; ++e) {
        const float* vrow = vbase + (size_t)e * LDQ;
        vb0[e] = (_Float16)vrow[d0];
        vb1[e] = (_Float16)vrow[d1];
      }
    }
    acc0 = wmma_f16(pa, vb0, acc0);
    acc1 = wmma_f16(pa, vb1, acc1);
  }

  // Finalize: out[row, h*DH + d] = acc / l  (only d < 24 stored).
#pragma unroll
  for (int e = 0; e < 8; ++e) {
    int row = q0 + e + (half << 3);
    float inv = 1.0f / lrow[e];
    int d0 = lane & 15;
    out[(size_t)row * GC + hh * GDH + d0] = acc0[e] * inv;
    int d1 = 16 + (lane & 15);
    if (d1 < GDH) out[(size_t)row * GC + hh * GDH + d1] = acc1[e] * inv;
  }
}

// ---------------------------------------------------------------------------
// GINE message + scatter: agg[dst] += relu(h[src] + edge_emb[edge_attr])
// One thread per (edge, channel); consecutive threads cover channels (coalesced).
// ---------------------------------------------------------------------------
__global__ void gine_scatter_kernel(const float* __restrict__ h,
                                    const int* __restrict__ edge_index,
                                    const int* __restrict__ edge_attr,
                                    const float* __restrict__ edge_emb,
                                    float* __restrict__ agg) {
  long long idx = (long long)blockIdx.x * blockDim.x + threadIdx.x;
  if (idx >= (long long)GE * GC) return;
  int e = (int)(idx / GC);
  int c = (int)(idx - (long long)e * GC);
  int src = edge_index[e];
  int dst = edge_index[GE + e];
  int et  = edge_attr[e];
  float m = h[(size_t)src * GC + c] + edge_emb[(size_t)et * GC + c];
  m = fmaxf(m, 0.f);
  atomicAdd(&agg[(size_t)dst * GC + c], m);
}

__global__ void zero_kernel(float* __restrict__ p, int n) {
  int i = blockIdx.x * blockDim.x + threadIdx.x;
  if (i < n) p[i] = 0.f;
}

__global__ void add_kernel(const float* __restrict__ a, const float* __restrict__ b,
                           float* __restrict__ o, int n) {
  int i = blockIdx.x * blockDim.x + threadIdx.x;
  if (i < n) o[i] = a[i] + b[i];
}

// ---------------------------------------------------------------------------
// BatchNorm1d training-mode: biased stats over N rows per channel.
// out = ((a (+ b)) - mean) * rsqrt(var + eps) * g + beta. One block per channel.
// ---------------------------------------------------------------------------
__global__ void batchnorm_kernel(const float* __restrict__ a,
                                 const float* __restrict__ b,
                                 const float* __restrict__ g,
                                 const float* __restrict__ beta,
                                 float* __restrict__ out) {
  const int c = blockIdx.x;
  __shared__ float ssum[256], ssq[256];
  float s = 0.f, sq = 0.f;
  for (int r = threadIdx.x; r < GN; r += blockDim.x) {
    float v = a[(size_t)r * GC + c];
    if (b) v += b[(size_t)r * GC + c];
    s += v; sq += v * v;
  }
  ssum[threadIdx.x] = s; ssq[threadIdx.x] = sq;
  __syncthreads();
  for (int st = 128; st > 0; st >>= 1) {
    if (threadIdx.x < st) {
      ssum[threadIdx.x] += ssum[threadIdx.x + st];
      ssq[threadIdx.x]  += ssq[threadIdx.x + st];
    }
    __syncthreads();
  }
  const float mean = ssum[0] * (1.0f / GN);
  const float var  = ssq[0] * (1.0f / GN) - mean * mean;
  const float inv  = rsqrtf(var + 1e-5f) * g[c];
  const float bb   = beta[c];
  for (int r = threadIdx.x; r < GN; r += blockDim.x) {
    float v = a[(size_t)r * GC + c];
    if (b) v += b[(size_t)r * GC + c];
    out[(size_t)r * GC + c] = (v - mean) * inv + bb;
  }
}

// LayerNorm over the last dim (C=96). One wave per row.
__global__ void layernorm_kernel(const float* __restrict__ y,
                                 const float* __restrict__ g,
                                 const float* __restrict__ b,
                                 float* __restrict__ out) {
  const int row = blockIdx.x;
  const int lane = threadIdx.x;
  float s = 0.f, sq = 0.f;
  for (int c = lane; c < GC; c += 32) {
    float v = y[(size_t)row * GC + c];
    s += v; sq += v * v;
  }
#pragma unroll
  for (int m = 1; m < 32; m <<= 1) { s += __shfl_xor(s, m, 32); sq += __shfl_xor(sq, m, 32); }
  const float mean = s * (1.0f / GC);
  const float var  = sq * (1.0f / GC) - mean * mean;
  const float inv  = rsqrtf(var + 1e-5f);
  for (int c = lane; c < GC; c += 32)
    out[(size_t)row * GC + c] = (y[(size_t)row * GC + c] - mean) * inv * g[c] + b[c];
}

// ---------------------------------------------------------------------------
extern "C" void kernel_launch(void* const* d_in, const int* in_sizes, int n_in,
                              void* d_out, int out_size, void* d_ws, size_t ws_size,
                              hipStream_t stream) {
  const float* x         = (const float*)d_in[0];
  const int*   edge_idx  = (const int*)d_in[1];
  const int*   edge_attr = (const int*)d_in[2];
  const float* w_in      = (const float*)d_in[3];
  const float* b_in      = (const float*)d_in[4];
  const float* edge_emb  = (const float*)d_in[5];
  const float* gine_w1   = (const float*)d_in[6];
  const float* gine_b1   = (const float*)d_in[7];
  const float* gine_w2   = (const float*)d_in[8];
  const float* gine_b2   = (const float*)d_in[9];
  const float* wqkv      = (const float*)d_in[10];
  const float* bqkv      = (const float*)d_in[11];
  const float* wo        = (const float*)d_in[12];
  const float* bo        = (const float*)d_in[13];
  const float* bn_g      = (const float*)d_in[14];
  const float* bn_b      = (const float*)d_in[15];
  const float* ffn_w1    = (const float*)d_in[16];
  const float* ffn_b1    = (const float*)d_in[17];
  const float* ffn_w2    = (const float*)d_in[18];
  const float* ffn_b2    = (const float*)d_in[19];
  const float* w_out     = (const float*)d_in[20];
  const float* b_out     = (const float*)d_in[21];
  const float* ln_g      = (const float*)d_in[22];
  const float* ln_b      = (const float*)d_in[23];

  const int N = GN, C = GC, IN = 64;
  const size_t NB = (size_t)N * C;

  float* ws = (float*)d_ws;
  float* h         = ws + 0 * NB;
  float* agg       = ws + 1 * NB;
  float* t0        = ws + 2 * NB;
  float* t1        = ws + 3 * NB;
  float* hl        = ws + 4 * NB;
  float* hlb       = ws + 5 * NB;
  float* qkv       = ws + 6 * NB;   // N x 3C  (3 NB)
  float* hattn_pre = ws + 9 * NB;
  float* hattn     = ws + 10 * NB;
  float* hattnb    = ws + 11 * NB;
  float* out0      = ws + 12 * NB;
  float* mid       = ws + 13 * NB;  // N x 2C  (2 NB)
  float* out1      = ws + 15 * NB;
  float* yb        = ws + 16 * NB;

  dim3 gblk(128);
  const int EW = 256;
  const int elemN = (int)NB;

  // h = x @ w_in + b_in
  gemm_wmma_kernel<false, false><<<dim3(N / 64, C / 16), gblk, 0, stream>>>(
      x, w_in, b_in, nullptr, h, N, IN, C);

  for (int i = 0; i < 2; ++i) {
    // ---- GINE local branch ----
    zero_kernel<<<(elemN + EW - 1) / EW, EW, 0, stream>>>(agg, elemN);
    {
      long long tot = (long long)GE * GC;
      int blocks = (int)((tot + EW - 1) / EW);
      gine_scatter_kernel<<<blocks, EW, 0, stream>>>(h, edge_idx, edge_attr, edge_emb, agg);
    }
    add_kernel<<<(elemN + EW - 1) / EW, EW, 0, stream>>>(h, agg, t0, elemN);
    gemm_wmma_kernel<true, false><<<dim3(N / 64, C / 16), gblk, 0, stream>>>(
        t0, gine_w1, gine_b1, nullptr, t1, N, C, C);
    gemm_wmma_kernel<false, false><<<dim3(N / 64, C / 16), gblk, 0, stream>>>(
        t1, gine_w2, gine_b2, nullptr, hl, N, C, C);
    batchnorm_kernel<<<C, 256, 0, stream>>>(hl, h, bn_g + (i * 3 + 0) * C,
                                            bn_b + (i * 3 + 0) * C, hlb);

    // ---- Attention branch ----
    gemm_wmma_kernel<false, false><<<dim3(N / 64, (3 * C) / 16), gblk, 0, stream>>>(
        h, wqkv + (size_t)i * C * 3 * C, bqkv + (size_t)i * 3 * C, nullptr,
        qkv, N, C, 3 * C);
    attention_kernel<<<dim3(N / 16, GHEADS), dim3(32), 0, stream>>>(qkv, hattn_pre);
    gemm_wmma_kernel<false, false><<<dim3(N / 64, C / 16), gblk, 0, stream>>>(
        hattn_pre, wo + (size_t)i * C * C, bo + (size_t)i * C, nullptr,
        hattn, N, C, C);
    batchnorm_kernel<<<C, 256, 0, stream>>>(hattn, h, bn_g + (i * 3 + 1) * C,
                                            bn_b + (i * 3 + 1) * C, hattnb);

    // ---- Merge + FFN ----
    add_kernel<<<(elemN + EW - 1) / EW, EW, 0, stream>>>(hlb, hattnb, out0, elemN);
    gemm_wmma_kernel<true, false><<<dim3(N / 64, (2 * C) / 16), gblk, 0, stream>>>(
        out0, ffn_w1 + (size_t)i * C * 2 * C, ffn_b1 + (size_t)i * 2 * C, nullptr,
        mid, N, C, 2 * C);
    gemm_wmma_kernel<false, true><<<dim3(N / 64, C / 16), gblk, 0, stream>>>(
        mid, ffn_w2 + (size_t)i * 2 * C * C, ffn_b2 + (size_t)i * C, out0,
        out1, N, 2 * C, C);
    batchnorm_kernel<<<C, 256, 0, stream>>>(out1, nullptr, bn_g + (i * 3 + 2) * C,
                                            bn_b + (i * 3 + 2) * C, h);
  }

  // y = h @ w_out + b_out ; out = layernorm(y)
  gemm_wmma_kernel<false, false><<<dim3(N / 64, C / 16), gblk, 0, stream>>>(
      h, w_out, b_out, nullptr, yb, N, C, C);
  layernorm_kernel<<<N, 32, 0, stream>>>(yb, ln_g, ln_b, (float*)d_out);
}
